// ContextAttnDecoder_53377853555210
// MI455X (gfx1250) — compile-verified, matrix operand order checked by reading, not run on hardware
//
#include <hip/hip_runtime.h>
#include <hip/hip_bf16.h>

// ---------------- problem constants ----------------
#define B_   128
#define T_   256
#define TC_  200
#define E_   256
#define H_   512
#define V_   50000
#define OOV_ 50
#define VEXT_ (V_ + OOV_)
#define H4_  (4 * H_)

typedef __attribute__((ext_vector_type(16))) __bf16 v16bf;
typedef __attribute__((ext_vector_type(8)))  __bf16 v8bf;
typedef __attribute__((ext_vector_type(8)))  float  v8f;

// =====================================================================
// GEMM:  C[128,N] (+)= A[128,K] * W[N,K]^T (+ bias[N])
// bf16 WMMA (v_wmma_f32_16x16x32_bf16), f32 accumulate.
//
// grid = (ceil(N/128), 8).  blockDim = 256 (8 waves).
//   mtile = blockIdx.y   (M = 128 -> 8 tiles)
//   wave w owns ntile = blockIdx.x*8 + w  -> W streamed exactly once.
// The 16xK A tile is staged once per block into LDS as bf16 and re-read
// by all 8 waves with 16B ds loads in the WMMA 16-bit A lane layout:
// lane(hi) reads k runs [k0+8*hi, +8) and [k0+16+8*hi, +8).
// K is a template constant (256/512) so the k-loop fully unrolls.
// =====================================================================
template<int K, bool ACC, bool BIAS>
__global__ void gemm_wmma_bf16(const float* __restrict__ A,
                               const float* __restrict__ W,
                               const float* __restrict__ bias,
                               float* __restrict__ C,
                               int N, int ldc)
{
    __shared__ __bf16 sA[16 * K];

    const int lane = threadIdx.x & 31;
    const int wave = threadIdx.x >> 5;
    const int mtile = blockIdx.y;

    // ---- cooperative A tile stage (f32 -> bf16), all 256 threads ----
    const float* __restrict__ Atile = A + (size_t)mtile * 16 * K;
#pragma unroll
    for (int i = threadIdx.x; i < 16 * K; i += 256)
        sA[i] = (__bf16)Atile[i];
    __syncthreads();

    const int ntile = blockIdx.x * 8 + wave;
    if (ntile * 16 >= N) return;                  // uniform per wave

    const int hi   = lane >> 4;
    const int ncol = ntile * 16 + (lane & 15);
    const float* __restrict__ wrow = W + (size_t)ncol * K;
    const __bf16* __restrict__ arow = sA + (lane & 15) * K;

    v8f acc = {};
#pragma unroll
    for (int k0 = 0; k0 < K; k0 += 32) {
        const float*  wp = wrow + k0 + hi * 8;
        const __bf16* ap = arow + k0 + hi * 8;
        __builtin_prefetch(wp + 512, 0, 1);       // global_prefetch_b8, 2KB ahead

        // A: two 16B LDS loads -> v16bf
        v8bf alo = *(const v8bf*)(ap);
        v8bf ahi = *(const v8bf*)(ap + 16);
        v16bf av = __builtin_shufflevector(alo, ahi,
                       0, 1, 2, 3, 4, 5, 6, 7, 8, 9, 10, 11, 12, 13, 14, 15);

        // W: four float4 global loads, convert to bf16
        float4 w0 = *(const float4*)(wp);
        float4 w1 = *(const float4*)(wp + 4);
        float4 w2 = *(const float4*)(wp + 16);
        float4 w3 = *(const float4*)(wp + 20);
        v16bf wv;
        wv[0]  = (__bf16)w0.x; wv[1]  = (__bf16)w0.y;
        wv[2]  = (__bf16)w0.z; wv[3]  = (__bf16)w0.w;
        wv[4]  = (__bf16)w1.x; wv[5]  = (__bf16)w1.y;
        wv[6]  = (__bf16)w1.z; wv[7]  = (__bf16)w1.w;
        wv[8]  = (__bf16)w2.x; wv[9]  = (__bf16)w2.y;
        wv[10] = (__bf16)w2.z; wv[11] = (__bf16)w2.w;
        wv[12] = (__bf16)w3.x; wv[13] = (__bf16)w3.y;
        wv[14] = (__bf16)w3.z; wv[15] = (__bf16)w3.w;

        acc = __builtin_amdgcn_wmma_f32_16x16x32_bf16(
                  false, av, false, wv, (short)0, acc, false, false);
    }

    float bv = 0.0f;
    if (BIAS) bv = bias[ncol];
#pragma unroll
    for (int r = 0; r < 8; ++r) {
        const int row = mtile * 16 + r + hi * 8;  // C/D layout: VGPR r -> M=r (+8 hi)
        const size_t idx = (size_t)row * ldc + ncol;
        float v = acc[r] + bv;
        if (ACC) v += C[idx];
        C[idx] = v;
    }
}

// =====================================================================
// Embedding gather: emb[b,:] = embedding[input[b],:]
// =====================================================================
__global__ void gather_emb_kernel(const int* __restrict__ input,
                                  const float* __restrict__ embedding,
                                  float* __restrict__ emb)
{
    int i = blockIdx.x * blockDim.x + threadIdx.x;   // B*E
    if (i >= B_ * E_) return;
    int b = i / E_, e = i - b * E_;
    emb[i] = embedding[(size_t)input[b] * E_ + e];
}

// =====================================================================
// LSTM cell (elementwise): gates[B,4H] + c0 -> h1, c1
// =====================================================================
__global__ void lstm_cell_kernel(const float* __restrict__ gates,
                                 const float* __restrict__ c0,
                                 float* __restrict__ h1,
                                 float* __restrict__ c1)
{
    int i = blockIdx.x * blockDim.x + threadIdx.x;   // B*H
    if (i >= B_ * H_) return;
    int b = i / H_, h = i - b * H_;
    const float* g = gates + (size_t)b * H4_;
    float gi = g[h], gf = g[H_ + h], gg = g[2 * H_ + h], go = g[3 * H_ + h];
    float si = 1.0f / (1.0f + __expf(-gi));
    float sf = 1.0f / (1.0f + __expf(-gf));
    float so = 1.0f / (1.0f + __expf(-go));
    float c  = sf * c0[i] + si * tanhf(gg);
    c1[i] = c;
    h1[i] = so * tanhf(c);
}

// =====================================================================
// Fused attention: scores = softmax_t(enc[b,t,:] . dec[b,:]);
//                  ctx[b,:] = sum_t scores[t]*enc[b,t,:]
// one block per batch row; wave-per-t dots with wave32 shfl reductions.
// =====================================================================
__global__ void attention_kernel(const float* __restrict__ enc,   // [B,T,H]
                                 const float* __restrict__ dec,   // [B,H]
                                 float* __restrict__ scores_out,  // [B,T]
                                 float* __restrict__ ctx_out,     // [B,H]
                                 int T)
{
    __shared__ float s_dec[H_];
    __shared__ float s_sc[T_];     // T <= 256
    __shared__ float red[16];

    const int b    = blockIdx.x;
    const int tid  = threadIdx.x;  // 256
    const int lane = tid & 31;
    const int wave = tid >> 5;     // 8 waves

    for (int h = tid; h < H_; h += 256) s_dec[h] = dec[(size_t)b * H_ + h];
    __syncthreads();

    const float* encb = enc + (size_t)b * T * H_;

    // ---- raw scores: one wave per t ----
    for (int t = wave; t < T; t += 8) {
        const float* row = encb + (size_t)t * H_;
        float s = 0.0f;
        for (int h = lane; h < H_; h += 32) s += row[h] * s_dec[h];
#pragma unroll
        for (int off = 16; off >= 1; off >>= 1) s += __shfl_xor(s, off, 32);
        if (lane == 0) s_sc[t] = s;
    }
    __syncthreads();

    // ---- softmax over T ----
    float m = -1e30f;
    for (int t = tid; t < T; t += 256) m = fmaxf(m, s_sc[t]);
#pragma unroll
    for (int off = 16; off >= 1; off >>= 1) m = fmaxf(m, __shfl_xor(m, off, 32));
    if (lane == 0) red[wave] = m;
    __syncthreads();
    if (wave == 0) {
        float mm = (lane < 8) ? red[lane] : -1e30f;
#pragma unroll
        for (int off = 16; off >= 1; off >>= 1) mm = fmaxf(mm, __shfl_xor(mm, off, 32));
        if (lane == 0) red[0] = mm;
    }
    __syncthreads();
    m = red[0];

    float ssum = 0.0f;
    for (int t = tid; t < T; t += 256) {
        float e = __expf(s_sc[t] - m);
        s_sc[t] = e;
        ssum += e;
    }
#pragma unroll
    for (int off = 16; off >= 1; off >>= 1) ssum += __shfl_xor(ssum, off, 32);
    if (lane == 0) red[8 + wave] = ssum;
    __syncthreads();
    if (wave == 0) {
        float t2 = (lane < 8) ? red[8 + lane] : 0.0f;
#pragma unroll
        for (int off = 16; off >= 1; off >>= 1) t2 += __shfl_xor(t2, off, 32);
        if (lane == 0) red[1] = t2;
    }
    __syncthreads();
    const float inv = 1.0f / red[1];

    for (int t = tid; t < T; t += 256) {
        float p = s_sc[t] * inv;
        s_sc[t] = p;
        scores_out[(size_t)b * T + t] = p;
    }
    __syncthreads();

    // ---- context: coalesced over h, reduce over t ----
    for (int h = tid; h < H_; h += 256) {
        float acc = 0.0f;
        for (int t = 0; t < T; ++t) acc += s_sc[t] * encb[(size_t)t * H_ + h];
        ctx_out[(size_t)b * H_ + h] = acc;
    }
}

// =====================================================================
// p_gen = sigmoid([ctx,cctx,h1,emb] . gen_W + gen_b + sig_b); 1.0 if no ctx
// =====================================================================
__global__ void pgen_kernel(const float* __restrict__ ctx,
                            const float* __restrict__ cctx,
                            const float* __restrict__ h1,
                            const float* __restrict__ emb,
                            const float* __restrict__ gen_W,
                            const float* __restrict__ gen_b,
                            const float* __restrict__ sig_b,
                            const int*   __restrict__ context_input,
                            float* __restrict__ p_gen)
{
    __shared__ float redf[8];
    __shared__ int   redi[8];
    const int b = blockIdx.x, tid = threadIdx.x, lane = tid & 31, wave = tid >> 5;
    const int FEAT = 3 * H_ + E_;   // 1792

    float s = 0.0f;
    for (int i = tid; i < FEAT; i += 256) {
        float f;
        if (i < H_)            f = ctx [(size_t)b * H_ + i];
        else if (i < 2 * H_)   f = cctx[(size_t)b * H_ + i - H_];
        else if (i < 3 * H_)   f = h1  [(size_t)b * H_ + i - 2 * H_];
        else                   f = emb [(size_t)b * E_ + i - 3 * H_];
        s += f * gen_W[i];
    }
    int cnt = 0;
    for (int t = tid; t < TC_; t += 256)
        cnt += (context_input[(size_t)b * TC_ + t] > 0) ? 1 : 0;

#pragma unroll
    for (int off = 16; off >= 1; off >>= 1) {
        s   += __shfl_xor(s, off, 32);
        cnt += __shfl_xor(cnt, off, 32);
    }
    if (lane == 0) { redf[wave] = s; redi[wave] = cnt; }
    __syncthreads();
    if (tid == 0) {
        float ss = 0.0f; int cc = 0;
        for (int w = 0; w < 8; ++w) { ss += redf[w]; cc += redi[w]; }
        float pg = 1.0f / (1.0f + __expf(-(ss + gen_b[0] + sig_b[0])));
        p_gen[b] = (cc == 0) ? 1.0f : pg;
    }
}

// =====================================================================
// p_copy: zero + scatter-add of cscores at context_input ids
// =====================================================================
__global__ void zero_kernel(float* __restrict__ p, size_t n)
{
    size_t stride = (size_t)gridDim.x * blockDim.x;
    for (size_t i = (size_t)blockIdx.x * blockDim.x + threadIdx.x; i < n; i += stride)
        p[i] = 0.0f;
}

__global__ void scatter_kernel(const int* __restrict__ context_input,
                               const float* __restrict__ cscores,
                               float* __restrict__ p_copy)
{
    int i = blockIdx.x * blockDim.x + threadIdx.x;   // B*Tc
    if (i >= B_ * TC_) return;
    int b = i / TC_;
    atomicAdd(&p_copy[(size_t)b * VEXT_ + context_input[i]], cscores[i]);
}

// =====================================================================
// Row max + sumexp over V logits (logits live in d_out, ldc = VEXT)
// =====================================================================
__global__ void rowreduce_kernel(const float* __restrict__ logits,  // [B,VEXT], first V valid
                                 float* __restrict__ rowmax,
                                 float* __restrict__ rowsum)
{
    __shared__ float red[16];
    const int b = blockIdx.x, tid = threadIdx.x, lane = tid & 31, wave = tid >> 5;
    const float* row = logits + (size_t)b * VEXT_;

    float m = -1e30f;
    for (int v = tid; v < V_; v += 256) m = fmaxf(m, row[v]);
#pragma unroll
    for (int off = 16; off >= 1; off >>= 1) m = fmaxf(m, __shfl_xor(m, off, 32));
    if (lane == 0) red[wave] = m;
    __syncthreads();
    if (wave == 0) {
        float mm = (lane < 8) ? red[lane] : -1e30f;
#pragma unroll
        for (int off = 16; off >= 1; off >>= 1) mm = fmaxf(mm, __shfl_xor(mm, off, 32));
        if (lane == 0) red[0] = mm;
    }
    __syncthreads();
    m = red[0];

    float s = 0.0f;
    for (int v = tid; v < V_; v += 256) s += __expf(row[v] - m);
#pragma unroll
    for (int off = 16; off >= 1; off >>= 1) s += __shfl_xor(s, off, 32);
    if (lane == 0) red[8 + wave] = s;
    __syncthreads();
    if (tid == 0) {
        float t = 0.0f;
        for (int w = 0; w < 8; ++w) t += red[8 + w];
        rowmax[b] = m;
        rowsum[b] = t;
    }
}

// =====================================================================
// Final: logp = log(clip(p_gen*softmax(logit) + (1-p_gen)*p_copy, 1e-10))
// in place over the logits region of d_out.
// =====================================================================
__global__ void final_kernel(float* __restrict__ logp,          // [B,VEXT] (holds logits)
                             const float* __restrict__ p_copy,  // [B,VEXT]
                             const float* __restrict__ p_gen,   // [B]
                             const float* __restrict__ rowmax,
                             const float* __restrict__ rowsum)
{
    size_t i = (size_t)blockIdx.x * blockDim.x + threadIdx.x;
    if (i >= (size_t)B_ * VEXT_) return;
    int b = (int)(i / VEXT_);
    int v = (int)(i - (size_t)b * VEXT_);
    float pg = p_gen[b];
    float pv = 0.0f;
    if (v < V_) pv = __expf(logp[i] - rowmax[b]) / rowsum[b];
    float p = pg * pv + (1.0f - pg) * p_copy[i];
    p = fmaxf(p, 1e-10f);
    logp[i] = __logf(p);
}

// =====================================================================
// Host side
// =====================================================================
extern "C" void kernel_launch(void* const* d_in, const int* in_sizes, int n_in,
                              void* d_out, int out_size, void* d_ws, size_t ws_size,
                              hipStream_t stream)
{
    const int*   input     = (const int*)  d_in[0];
    const float* h0        = (const float*)d_in[1];
    const float* c0        = (const float*)d_in[2];
    const float* enc       = (const float*)d_in[3];
    const float* cenc      = (const float*)d_in[4];
    const int*   ctx_in    = (const int*)  d_in[5];
    const float* embedding = (const float*)d_in[6];
    const float* W_ih      = (const float*)d_in[7];
    const float* W_hh      = (const float*)d_in[8];
    const float* b_ih      = (const float*)d_in[9];
    const float* b_hh      = (const float*)d_in[10];
    const float* attn_W    = (const float*)d_in[11];
    const float* attn_b    = (const float*)d_in[12];
    const float* cattn_W   = (const float*)d_in[13];
    const float* cattn_b   = (const float*)d_in[14];
    const float* gen_W     = (const float*)d_in[15];
    const float* gen_b     = (const float*)d_in[16];
    const float* sig_b     = (const float*)d_in[17];
    const float* out_W     = (const float*)d_in[18];
    const float* out_b     = (const float*)d_in[19];

    float* out_logp = (float*)d_out;                       // [B, VEXT]
    float* out_h1   = out_logp + (size_t)B_ * VEXT_;       // [B, H]
    float* out_c1   = out_h1   + (size_t)B_ * H_;          // [B, H]

    // ---- workspace carve-up (floats) ----
    float* ws       = (float*)d_ws;
    float* emb_ws   = ws;                  ws += (size_t)B_ * E_;
    float* gates_ws = ws;                  ws += (size_t)B_ * H4_;
    float* dec_ws   = ws;                  ws += (size_t)B_ * H_;
    float* cdec_ws  = ws;                  ws += (size_t)B_ * H_;
    float* scores_ws  = ws;                ws += (size_t)B_ * T_;
    float* cscores_ws = ws;                ws += (size_t)B_ * TC_;
    float* ctx_ws   = ws;                  ws += (size_t)B_ * H_;
    float* cctx_ws  = ws;                  ws += (size_t)B_ * H_;
    float* pgen_ws  = ws;                  ws += B_;
    float* rowmax_ws = ws;                 ws += B_;
    float* rowsum_ws = ws;                 ws += B_;
    float* pcopy_ws  = ws;                 ws += (size_t)B_ * VEXT_;

    // grid helper: (ceil(ntiles/8), 8 mtiles)
    auto gemm_grid = [](int N) { return dim3((N / 16 + 7) / 8, 8); };

    // 1. embedding gather
    gather_emb_kernel<<<(B_ * E_ + 255) / 256, 256, 0, stream>>>(input, embedding, emb_ws);

    // 2-3. LSTM gates = emb@W_ih^T + b_ih  (then += h0@W_hh^T + b_hh)
    gemm_wmma_bf16<E_, false, true><<<gemm_grid(H4_), 256, 0, stream>>>(
        emb_ws, W_ih, b_ih, gates_ws, H4_, H4_);
    gemm_wmma_bf16<H_, true, true><<<gemm_grid(H4_), 256, 0, stream>>>(
        h0, W_hh, b_hh, gates_ws, H4_, H4_);

    // 4. LSTM cell -> h1, c1 (straight into d_out tail)
    lstm_cell_kernel<<<(B_ * H_ + 255) / 256, 256, 0, stream>>>(gates_ws, c0, out_h1, out_c1);

    // 5-6. attention projections
    gemm_wmma_bf16<H_, false, true><<<gemm_grid(H_), 256, 0, stream>>>(
        out_h1, attn_W, attn_b, dec_ws, H_, H_);
    gemm_wmma_bf16<H_, false, true><<<gemm_grid(H_), 256, 0, stream>>>(
        out_h1, cattn_W, cattn_b, cdec_ws, H_, H_);

    // 7-8. fused score/softmax/context (encoder + context-encoder)
    attention_kernel<<<B_, 256, 0, stream>>>(enc,  dec_ws,  scores_ws,  ctx_ws,  T_);
    attention_kernel<<<B_, 256, 0, stream>>>(cenc, cdec_ws, cscores_ws, cctx_ws, TC_);

    // 9. p_gen
    pgen_kernel<<<B_, 256, 0, stream>>>(ctx_ws, cctx_ws, out_h1, emb_ws,
                                        gen_W, gen_b, sig_b, ctx_in, pgen_ws);

    // 10-11. p_copy = scatter(cscores)
    zero_kernel<<<4096, 256, 0, stream>>>(pcopy_ws, (size_t)B_ * VEXT_);
    scatter_kernel<<<(B_ * TC_ + 255) / 256, 256, 0, stream>>>(ctx_in, cscores_ws, pcopy_ws);

    // 12. vocab logits -> d_out logp area (ldc = VEXT), the dominant GEMM
    gemm_wmma_bf16<H_, false, true><<<gemm_grid(V_), 256, 0, stream>>>(
        out_h1, out_W, out_b, out_logp, V_, VEXT_);

    // 13. softmax row statistics
    rowreduce_kernel<<<B_, 256, 0, stream>>>(out_logp, rowmax_ws, rowsum_ws);

    // 14. final mix + log, in place
    final_kernel<<<(int)(((size_t)B_ * VEXT_ + 255) / 256), 256, 0, stream>>>(
        out_logp, pcopy_ws, pgen_ws, rowmax_ws, rowsum_ws);
}
